// GCNLayerNonNeighb_38388417692550
// MI455X (gfx1250) — matrix operand order repaired and checked by reading, not compiled, not run on hardware
//
#include <hip/hip_runtime.h>
#include <hip/hip_bf16.h>

typedef __attribute__((ext_vector_type(16))) _Float16 v16h;
typedef __attribute__((ext_vector_type(8)))  _Float16 v8h;
typedef __attribute__((ext_vector_type(8)))  float    v8f;
typedef __attribute__((ext_vector_type(4)))  float    v4f;

#define D 128

// ---------------- workspace init: zero neigh_sum + total ----------------
__global__ void zero_ws_kernel(float* __restrict__ nsum, float* __restrict__ total, long n4) {
  long i = (long)blockIdx.x * blockDim.x + threadIdx.x;
  float4 z = make_float4(0.f, 0.f, 0.f, 0.f);
  for (long j = i; j < n4; j += (long)gridDim.x * blockDim.x)
    ((float4*)nsum)[j] = z;
  if (i < D) total[i] = 0.f;
}

// ---------------- convert W (fp32) -> f16 for WMMA B operand ----------------
__global__ void convW_kernel(const float* __restrict__ W, _Float16* __restrict__ Wh) {
  int i = blockIdx.x * blockDim.x + threadIdx.x;
  if (i < D * D) Wh[i] = (_Float16)W[i];
}

// ---------------- column totals: total[c] = sum_r feat[r][c] ----------------
__global__ void col_total_kernel(const float* __restrict__ feat,
                                 float* __restrict__ total, int N) {
  int c = threadIdx.x;              // 0..127, one column per thread
  float acc = 0.f;
  for (int r = blockIdx.x; r < N; r += gridDim.x)
    acc += feat[(long)r * D + c];
  atomicAdd(&total[c], acc);        // 128 atomics per block
}

// ---------------- edge scatter: nsum[dst] += feat[src], one wave32/edge ----
__global__ void scatter_kernel(const float* __restrict__ feat,
                               const int* __restrict__ src,
                               const int* __restrict__ dst,
                               float* __restrict__ nsum, int E) {
  long gtid = (long)blockIdx.x * blockDim.x + threadIdx.x;
  int  e    = (int)(gtid >> 5);     // wave index == edge index
  int  lane = threadIdx.x & 31;
  if (e >= E) return;
  // e / src / dst are wave-uniform: force to SGPRs so the index fetches become
  // scalar loads and the row accesses use saddr + lane-offset addressing.
  e = __builtin_amdgcn_readfirstlane(e);
  int s = __builtin_amdgcn_readfirstlane(src[e]);
  int d = __builtin_amdgcn_readfirstlane(dst[e]);
  // coalesced 512B row gather: lane covers 4 consecutive floats (hot in L2)
  float4 v = ((const float4*)(feat + (long)s * D))[lane];
  float* o = nsum + (long)d * D + (lane << 2);
  atomicAdd(o + 0, v.x);
  atomicAdd(o + 1, v.y);
  atomicAdd(o + 2, v.z);
  atomicAdd(o + 3, v.w);
}

// ---------------- fused: h = total - nsum ; L2-normalize ; out = h @ W^T + b
// Block = 256 threads (8 waves). Block owns 16 output rows; each wave owns one
// 16-wide N-tile. GEMM via V_WMMA_F32_16X16X32_F16, K=128 in 4 steps.
__global__ void __launch_bounds__(256, 2)
fused_norm_gemm_kernel(const float* __restrict__ nsum,
                       const float* __restrict__ total,
                       const _Float16* __restrict__ Wh,
                       const float* __restrict__ bias,
                       float* __restrict__ out, int N) {
  __shared__ __align__(16) _Float16 h16[16][D];  // normalized f16 A tile, row-major
  __shared__ float psum[16][17];                  // per-row sum-of-squares partials
  __shared__ float rinv[16];

  const int t    = threadIdx.x;
  const int row0 = blockIdx.x * 16;

  // ---- phase 1: build h tile in registers (no divergence; clamp OOB rows) --
  const int r = t >> 4;           // 0..15  (local row)
  const int c = (t & 15) * 8;     // 0,8,...,120 (col group of 8)
  int row = row0 + r;
  if (row >= N) row = N - 1;      // clamp instead of branch-out (keeps EXEC full)

  // nsum is consumed exactly once -> non-temporal (don't evict hot L2 lines)
  const v4f n0 = __builtin_nontemporal_load((const v4f*)(nsum + (long)row * D + c));
  const v4f n1 = __builtin_nontemporal_load((const v4f*)(nsum + (long)row * D + c + 4));
  const v4f t0 = *(const v4f*)(total + c);
  const v4f t1 = *(const v4f*)(total + c + 4);
  float h[8];
  #pragma unroll
  for (int i = 0; i < 4; ++i) { h[i] = t0[i] - n0[i]; h[4 + i] = t1[i] - n1[i]; }

  float ss = 0.f;
  #pragma unroll
  for (int i = 0; i < 8; ++i) ss += h[i] * h[i];
  psum[r][t & 15] = ss;
  __syncthreads();

  if (t < 16) {
    float s = 0.f;
    #pragma unroll
    for (int i = 0; i < 16; ++i) s += psum[t][i];
    rinv[t] = 1.0f / fmaxf(sqrtf(s), 1e-12f);
  }
  __syncthreads();

  const float ri = rinv[r];
  #pragma unroll
  for (int i = 0; i < 8; ++i) h16[r][c + i] = (_Float16)(h[i] * ri);
  __syncthreads();

  // ---- phase 2: WMMA GEMM. wave wv -> N-tile [wv*16, wv*16+16) --------------
  const int wv   = t >> 5;        // 0..7
  const int lane = t & 31;
  const int nl   = lane & 15;     // local M (for A) / local N (for B,C)
  const int hi   = lane >> 4;     // half-wave selector

  const _Float16* wrow = Wh + (long)(wv * 16 + nl) * D;  // B: lane = N -> W row

  // fold bias into the accumulator init: C layout has N = nl for every VGPR,
  // so all 8 accumulator regs start at bias[n]. Stores then depend only on
  // the final WMMA (no post-add chain).
  const float bn = bias[wv * 16 + nl];
  v8f acc;
  #pragma unroll
  for (int g = 0; g < 8; ++g) acc[g] = bn;

  #pragma unroll
  for (int kt = 0; kt < 4; ++kt) {
    union { v16h v; v8h p[2]; } af, bf;
    // A fragment (16x32 f16): lanes 0-15 K in {base..base+7, base+16..base+23},
    // lanes 16-31 shifted by 8 (ISA 7.12.2 16-bit A layout). Two 16B LDS loads.
    const int abase = kt * 32 + hi * 8;
    af.p[0] = *(const v8h*)&h16[nl][abase];
    af.p[1] = *(const v8h*)&h16[nl][abase + 16];
    // B fragment (32x16 f16): lane = N, same K layout as A (two 16B global loads).
    const int bbase = kt * 32 + hi * 8;
    bf.p[0] = *(const v8h*)(wrow + bbase);
    bf.p[1] = *(const v8h*)(wrow + bbase + 16);
    acc = __builtin_amdgcn_wmma_f32_16x16x32_f16(
        /*neg_a=*/false, af.v, /*neg_b=*/false, bf.v,
        /*c_mod=*/(short)0, acc, /*reuse_a=*/false, /*reuse_b=*/false);
  }

  // ---- phase 3: store. C/D layout: VGPR g -> M = g + hi*8, N = nl ----------
  // Block-uniform full-tile test: only the last block can be partial, so the
  // hot path is 8 unguarded non-temporal stores (single s_cbranch, no exec
  // toggling per element).
  float* obase = out + (long)(row0 + hi * 8) * D + wv * 16 + nl;
  if (row0 + 16 <= N) {
    #pragma unroll
    for (int g = 0; g < 8; ++g)
      __builtin_nontemporal_store(acc[g], obase + (long)g * D);
  } else {
    #pragma unroll
    for (int g = 0; g < 8; ++g) {
      int m = row0 + hi * 8 + g;
      if (m < N) out[(long)m * D + wv * 16 + nl] = acc[g];
    }
  }
}

extern "C" void kernel_launch(void* const* d_in, const int* in_sizes, int n_in,
                              void* d_out, int out_size, void* d_ws, size_t ws_size,
                              hipStream_t stream) {
  const float* feat = (const float*)d_in[0];   // [N,128] fp32
  const float* W    = (const float*)d_in[1];   // [128,128] fp32
  const float* bias = (const float*)d_in[2];   // [128] fp32
  const int*   src  = (const int*)d_in[3];     // [E] int32
  const int*   dst  = (const int*)d_in[4];     // [E] int32
  float*       out  = (float*)d_out;           // [N,128] fp32

  const int N = in_sizes[0] / D;
  const int E = in_sizes[3];

  // workspace layout: [ nsum: N*D f32 | total: D f32 | pad | Wh: D*D f16 ]
  char* ws = (char*)d_ws;
  float*    nsum  = (float*)ws;
  float*    total = (float*)(ws + (size_t)N * D * sizeof(float));
  _Float16* Wh    = (_Float16*)(ws + (size_t)N * D * sizeof(float) + 512);

  const long n4 = (long)N * D / 4;
  zero_ws_kernel<<<2048, 256, 0, stream>>>(nsum, total, n4);
  convW_kernel<<<(D * D + 255) / 256, 256, 0, stream>>>(W, Wh);
  col_total_kernel<<<512, D, 0, stream>>>(feat, total, N);

  const int blocksE = (E + 7) / 8;   // 8 wave32 (edges) per 256-thread block
  scatter_kernel<<<blocksE, 256, 0, stream>>>(feat, src, dst, nsum, E);

  const int blocksN = (N + 15) / 16; // 16 output rows per block
  fused_norm_gemm_kernel<<<blocksN, 256, 0, stream>>>(nsum, total, Wh, bias, out, N);
}
